// SpikingClassifier_13022340841689
// MI455X (gfx1250) — matrix operand order repaired and checked by reading, not compiled, not run on hardware
//
#include <hip/hip_runtime.h>

// ---------------------------------------------------------------------------
// SpikingClassifier for MI455X (gfx1250, wave32, WMMA 16x16x32 bf16)
//
// Structure exploited: W_lat = -0.1*(ones - eye)  =>  v_h @ W_lat collapses
// to -0.1*(rowsum(v_h) - v_h)  (removes 343 GFLOP of GEMM work, exactly).
//
// GEMM1 (x @ W_in^T): bf16 hi/lo split, 3 WMMA products (near-fp32 accuracy,
// needed because the spike threshold is discontinuous).
// GEMM2 (6 stacked [spikes x5, relu(I_in)] @ W_out^T): spikes are exact in
// bf16, so A is single-plane; W_out is hi/lo (2 WMMA products).
// ---------------------------------------------------------------------------

typedef __attribute__((ext_vector_type(16))) __bf16 v16bf;
typedef __attribute__((ext_vector_type(8)))  float  v8f;

#define DEVINL static __device__ __forceinline__

// ---- constants of the problem ----
#define BATCH   2048
#define N_IN    4096
#define N_HID   4096
#define N_CLS   1024
#define N_STEPS 5
#define M_BIG   (6 * BATCH)   // 5 spike blocks + 1 relu(I_in) block = 12288

// ---- bf16 helpers (RNE) ----
DEVINL unsigned short f2bf(float f) {
    union { float f; unsigned int u; } c; c.f = f;
    unsigned int u = c.u;
    u += 0x7FFFu + ((u >> 16) & 1u);            // round-to-nearest-even
    return (unsigned short)(u >> 16);
}
DEVINL float bf2f(unsigned short h) {
    union { unsigned int u; float f; } c; c.u = ((unsigned int)h) << 16;
    return c.f;
}

// ---- WMMA fragment loads --------------------------------------------------
// A (16x32 bf16, MxK): lane L: m = L&15, kh = L>>4.
//   elems 0..7  = A[m][kt + kh*8 + 0..7]
//   elems 8..15 = A[m][kt + 16 + kh*8 + 0..7]
// B (32x16 bf16, KxN): lane L: n = L&15, kh = L>>4.
//   elems 0..7  = B_rowmajorNK[n][kt + kh*8 + 0..7], elems 8..15 = B_rowmajorNK[n][kt + 16 + kh*8 + 0..7]   (i.e. W[n][k])
union FragU { v16bf v; uint4 q[2]; };

DEVINL v16bf ldfragA(const unsigned short* p) {      // p -> elem0, second chunk at +16 elems
    FragU u;
    u.q[0] = *(const uint4*)(p);
    u.q[1] = *(const uint4*)(p + 16);
    return u.v;
}
DEVINL v16bf ldfragB(const unsigned short* p) {      // p -> elem0, second chunk at +16 elems
    FragU u;
    u.q[0] = *(const uint4*)(p);
    u.q[1] = *(const uint4*)(p + 16);
    return u.v;
}

DEVINL v8f wmma_bf16(v16bf a, v16bf b, v8f c) {
    return __builtin_amdgcn_wmma_f32_16x16x32_bf16(
        /*neg_a=*/false, a, /*neg_b=*/false, b,
        /*c_mod=*/(short)0, c, /*reuse_a=*/false, /*reuse_b=*/false);
}

// ---------------------------------------------------------------------------
// Split f32 -> (bf16 hi, bf16 lo) planes.  n4 = n/4 float4 units.
// ---------------------------------------------------------------------------
__global__ __launch_bounds__(256) void split_bf16_kernel(
    const float* __restrict__ src,
    unsigned short* __restrict__ hi,
    unsigned short* __restrict__ lo,
    int n4)
{
    int i = blockIdx.x * 256 + threadIdx.x;
    if (i >= n4) return;
    float4 x = ((const float4*)src)[i];
    unsigned short h0 = f2bf(x.x), h1 = f2bf(x.y), h2 = f2bf(x.z), h3 = f2bf(x.w);
    unsigned short l0 = f2bf(x.x - bf2f(h0));
    unsigned short l1 = f2bf(x.y - bf2f(h1));
    unsigned short l2 = f2bf(x.z - bf2f(h2));
    unsigned short l3 = f2bf(x.w - bf2f(h3));
    uint2 ph = { (unsigned)h0 | ((unsigned)h1 << 16), (unsigned)h2 | ((unsigned)h3 << 16) };
    uint2 pl = { (unsigned)l0 | ((unsigned)l1 << 16), (unsigned)l2 | ((unsigned)l3 << 16) };
    ((uint2*)hi)[i] = ph;
    ((uint2*)lo)[i] = pl;
}

// ---------------------------------------------------------------------------
// TN GEMM: C[m][n] = sum_k A[m][k] * B[n][k]  (+ bias[n])
// A: M x K bf16 plane(s), B: N x K bf16 plane(s), C: M x N f32.
// Block = 128 threads = 4 waves (2x2), block tile 128x128, wave tile 64x64.
// NPROD: 3 -> Ahi*Bhi + Ahi*Blo + Alo*Bhi  (emulated fp32)
//        2 -> Ahi*Bhi + Ahi*Blo            (A exact in bf16)
// M, N, K must be multiples of 128/128/32 (true for all calls here).
// ---------------------------------------------------------------------------
template <int NPROD>
__global__ __launch_bounds__(128) void gemm_wmma_kernel(
    const unsigned short* __restrict__ Ahi,
    const unsigned short* __restrict__ Alo,
    const unsigned short* __restrict__ Bhi,
    const unsigned short* __restrict__ Blo,
    const float* __restrict__ bias,
    float* __restrict__ C,
    int N, int K)
{
    const int tid  = threadIdx.x;
    const int wave = tid >> 5;
    const int lane = tid & 31;
    const int lm   = lane & 15;
    const int lh   = lane >> 4;

    const int M0 = blockIdx.y * 128 + (wave >> 1) * 64;
    const int N0 = blockIdx.x * 128 + (wave & 1) * 64;

    v8f acc[4][4];
    const v8f vzero = { 0.f, 0.f, 0.f, 0.f, 0.f, 0.f, 0.f, 0.f };
#pragma unroll
    for (int i = 0; i < 4; ++i)
#pragma unroll
        for (int j = 0; j < 4; ++j) acc[i][j] = vzero;

    // per-lane base pointers
    const unsigned short* aHiP[4];
    const unsigned short* aLoP[4];
    const unsigned short* bHiP[4];
    const unsigned short* bLoP[4];
#pragma unroll
    for (int i = 0; i < 4; ++i) {
        size_t off = (size_t)(M0 + i * 16 + lm) * (size_t)K + (size_t)(lh * 8);
        aHiP[i] = Ahi + off;
        aLoP[i] = (NPROD == 3) ? (Alo + off) : Ahi + off;   // unused when NPROD<3
    }
#pragma unroll
    for (int j = 0; j < 4; ++j) {
        size_t off = (size_t)(N0 + j * 16 + lm) * (size_t)K + (size_t)(lh * 8);
        bHiP[j] = Bhi + off;
        bLoP[j] = (NPROD >= 2) ? (Blo + off) : Bhi + off;   // unused when NPROD<2
    }

    for (int kt = 0; kt < K; kt += 32) {
        v16bf bh[4], bl[4];
#pragma unroll
        for (int j = 0; j < 4; ++j) {
            bh[j] = ldfragB(bHiP[j] + kt);
            if (NPROD >= 2) bl[j] = ldfragB(bLoP[j] + kt);
        }
#pragma unroll
        for (int i = 0; i < 4; ++i) {
            v16bf ah = ldfragA(aHiP[i] + kt);
            v16bf al;
            if (NPROD == 3) al = ldfragA(aLoP[i] + kt);
#pragma unroll
            for (int j = 0; j < 4; ++j) {
                acc[i][j] = wmma_bf16(ah, bh[j], acc[i][j]);
                if (NPROD >= 2) acc[i][j] = wmma_bf16(ah, bl[j], acc[i][j]);
                if (NPROD == 3) acc[i][j] = wmma_bf16(al, bh[j], acc[i][j]);
            }
        }
    }

    // store: C/D layout: VGPR r, lane L -> C[sub_m + (L>>4)*8 + r][sub_n + (L&15)]
#pragma unroll
    for (int i = 0; i < 4; ++i) {
        const int mb = M0 + i * 16 + lh * 8;
#pragma unroll
        for (int j = 0; j < 4; ++j) {
            const int n  = N0 + j * 16 + lm;
            const float bv = bias ? bias[n] : 0.0f;
            float* cp = C + (size_t)mb * (size_t)N + (size_t)n;
#pragma unroll
            for (int r = 0; r < 8; ++r)
                cp[(size_t)r * (size_t)N] = acc[i][j][r] + bv;
        }
    }
}

// ---------------------------------------------------------------------------
// Hidden-layer spiking dynamics. One block (1024 thr) per batch row.
// Each thread owns 4 hidden units.  I_lat = -0.1*(rowsum(v) - v).
// Writes spike blocks t=0..4 and relu(I_in) block (t=5) of A_big (bf16).
// ---------------------------------------------------------------------------
__global__ __launch_bounds__(1024) void snn_hidden_kernel(
    const float* __restrict__ I_in,           // [BATCH, N_HID]
    unsigned short* __restrict__ Abig)        // [M_BIG, N_HID] bf16
{
    __shared__ float red[1024];
    const int b   = blockIdx.x;
    const int tid = threadIdx.x;

    float4 Ii = ((const float4*)(I_in + (size_t)b * N_HID))[tid];
    const float Iv[4] = { Ii.x, Ii.y, Ii.z, Ii.w };
    float v[4] = { 0.f, 0.f, 0.f, 0.f };

    for (int t = 0; t < N_STEPS; ++t) {
        red[tid] = v[0] + v[1] + v[2] + v[3];
        __syncthreads();
        for (int off = 512; off > 0; off >>= 1) {
            if (tid < off) red[tid] += red[tid + off];
            __syncthreads();
        }
        const float S = red[0];
        __syncthreads();

        unsigned short sp[4];
#pragma unroll
        for (int c = 0; c < 4; ++c) {
            float vn = 0.8f * v[c] + 0.2f * (Iv[c] - 0.1f * (S - v[c]));
            if (vn >= 0.5f) { sp[c] = 0x3F80; v[c] = 0.0f; }   // bf16(1.0)
            else            { sp[c] = 0;      v[c] = vn;   }
        }
        uint2 pk = { (unsigned)sp[0] | ((unsigned)sp[1] << 16),
                     (unsigned)sp[2] | ((unsigned)sp[3] << 16) };
        ((uint2*)(Abig + ((size_t)t * BATCH + b) * N_HID))[tid] = pk;
    }

    // block 5: relu(I_in) in bf16 (feeds the analog path)
    unsigned short r0 = f2bf(fmaxf(Iv[0], 0.f));
    unsigned short r1 = f2bf(fmaxf(Iv[1], 0.f));
    unsigned short r2 = f2bf(fmaxf(Iv[2], 0.f));
    unsigned short r3 = f2bf(fmaxf(Iv[3], 0.f));
    uint2 pr = { (unsigned)r0 | ((unsigned)r1 << 16),
                 (unsigned)r2 | ((unsigned)r3 << 16) };
    ((uint2*)(Abig + ((size_t)N_STEPS * BATCH + b) * N_HID))[tid] = pr;
}

// ---------------------------------------------------------------------------
// Output-layer dynamics + final combine.  One thread per (b, c).
// Obig rows: t*BATCH+b -> I_out at step t (bias included); 5*BATCH+b -> analog.
// ---------------------------------------------------------------------------
__global__ __launch_bounds__(256) void snn_output_kernel(
    const float* __restrict__ Obig,           // [M_BIG, N_CLS]
    float* __restrict__ out)                  // [BATCH, N_CLS]
{
    const int idx = blockIdx.x * 256 + threadIdx.x;   // < BATCH*N_CLS
    const int b = idx >> 10;          // / N_CLS
    const int c = idx & (N_CLS - 1);

    float v = 0.f, counts = 0.f;
#pragma unroll
    for (int t = 0; t < N_STEPS; ++t) {
        float I = Obig[((size_t)t * BATCH + b) * N_CLS + c];
        v = 0.8f * v + 0.2f * I;
        if (v >= 0.5f) { counts += 1.0f; v = 0.0f; }
    }
    float analog = Obig[((size_t)N_STEPS * BATCH + b) * N_CLS + c];
    out[idx] = counts / 5.0f + 0.5f * analog;
}

// ---------------------------------------------------------------------------
// Host-side orchestration
// ---------------------------------------------------------------------------
extern "C" void kernel_launch(void* const* d_in, const int* in_sizes, int n_in,
                              void* d_out, int out_size, void* d_ws, size_t ws_size,
                              hipStream_t stream) {
    (void)in_sizes; (void)n_in; (void)out_size; (void)ws_size;

    const float* x       = (const float*)d_in[0];   // [B, N_IN]
    const float* W_in_w  = (const float*)d_in[1];   // [N_HID, N_IN]
    const float* W_in_b  = (const float*)d_in[2];   // [N_HID]
    const float* W_out_w = (const float*)d_in[3];   // [N_CLS, N_HID]
    const float* W_out_b = (const float*)d_in[4];   // [N_CLS]
    // d_in[5] = W_lat: known structure -0.1*(ones-eye), folded analytically.

    // workspace carve-up (all sizes are multiples of 256B)
    char* ws = (char*)d_ws;
    size_t off = 0;
    auto take = [&](size_t bytes) { char* p = ws + off; off += bytes; return p; };

    unsigned short* xhi   = (unsigned short*)take((size_t)BATCH * N_IN  * 2);
    unsigned short* xlo   = (unsigned short*)take((size_t)BATCH * N_IN  * 2);
    unsigned short* winhi = (unsigned short*)take((size_t)N_HID * N_IN  * 2);
    unsigned short* winlo = (unsigned short*)take((size_t)N_HID * N_IN  * 2);
    unsigned short* wouthi= (unsigned short*)take((size_t)N_CLS * N_HID * 2);
    unsigned short* woutlo= (unsigned short*)take((size_t)N_CLS * N_HID * 2);
    float*          Iin   = (float*)         take((size_t)BATCH * N_HID * 4);
    unsigned short* Abig  = (unsigned short*)take((size_t)M_BIG * N_HID * 2);
    float*          Obig  = (float*)         take((size_t)M_BIG * N_CLS * 4);

    // 1) f32 -> bf16 hi/lo planes
    {
        int n4 = BATCH * N_IN / 4;
        split_bf16_kernel<<<(n4 + 255) / 256, 256, 0, stream>>>(x, xhi, xlo, n4);
        n4 = N_HID * N_IN / 4;
        split_bf16_kernel<<<(n4 + 255) / 256, 256, 0, stream>>>(W_in_w, winhi, winlo, n4);
        n4 = N_CLS * N_HID / 4;
        split_bf16_kernel<<<(n4 + 255) / 256, 256, 0, stream>>>(W_out_w, wouthi, woutlo, n4);
    }

    // 2) I_in = x @ W_in^T + b   (3-product split-bf16 WMMA, near-fp32)
    gemm_wmma_kernel<3><<<dim3(N_HID / 128, BATCH / 128), 128, 0, stream>>>(
        xhi, xlo, winhi, winlo, W_in_b, Iin, N_HID, N_IN);

    // 3) hidden spiking dynamics -> A_big (5 spike blocks + relu block)
    snn_hidden_kernel<<<BATCH, 1024, 0, stream>>>(Iin, Abig);

    // 4) Out_big = A_big @ W_out^T + b   (A exact in bf16; W_out hi/lo)
    gemm_wmma_kernel<2><<<dim3(N_CLS / 128, M_BIG / 128), 128, 0, stream>>>(
        Abig, nullptr, wouthi, woutlo, W_out_b, Obig, N_CLS, N_HID);

    // 5) output neuron dynamics + combine
    snn_output_kernel<<<(BATCH * N_CLS) / 256, 256, 0, stream>>>(Obig, (float*)d_out);
}
